// FlashCoreAttention_50251117363623
// MI455X (gfx1250) — compile-verified, matrix operand order checked by reading, not run on hardware
//
#include <hip/hip_runtime.h>

// ---------------------------------------------------------------------------
// FlashCoreAttention for MI455X (gfx1250, wave32, WMMA bf16 16x16x32)
// T=1024, B=4, H=16, D=64.  q/k/v: [T,B,H*D] fp32.  mask: [T,T,B*H] fp32.
// Grid: (qt=64, b=4). Block: 512 threads = 16 waves; wave w = head w.
// Mask stream (268 MB) is the roofline bottleneck: staged block-wide into LDS
// with a VGPR double buffer so global loads overlap WMMA/softmax compute.
// Row-sums computed by an extra WMMA against an all-ones B matrix; row-max
// reduced with DPP16 butterflies (no LDS traffic).
// ---------------------------------------------------------------------------

#define T_LEN   1024
#define ROW_STR 4096          // floats per t-row in q/k/v/out: B * H * D
#define MASK_QS 65536         // floats per tgt row in mask: T * B*H
#define NORM_F  0.125f        // 1/sqrt(64)

typedef __bf16 v16bf __attribute__((ext_vector_type(16)));
typedef __bf16 v8bf  __attribute__((ext_vector_type(8)));
typedef float  v8f   __attribute__((ext_vector_type(8)));

// K-offset of element j of a 16-bit A/B fragment (16x32), given lane half.
__device__ __forceinline__ int koff(int hf, int j) {
    return (j < 8) ? (8 * hf + j) : (16 + 8 * hf + (j - 8));
}

// One DPP16 butterfly step folded into a max (row ops stay within 16 lanes).
#define DPP_MAXSTEP(x, ctrl)                                                   \
    (x) = fmaxf((x), __int_as_float(__builtin_amdgcn_update_dpp(               \
                    __float_as_int(x), __float_as_int(x), (ctrl), 0xf, 0xf, true)))

// Reduce max across the 16-lane row group: xor1, xor2, mirror8, mirror16.
__device__ __forceinline__ float rowmax16(float x) {
    DPP_MAXSTEP(x, 0xB1);   // quad_perm(1,0,3,2)  : xor 1
    DPP_MAXSTEP(x, 0x4E);   // quad_perm(2,3,0,1)  : xor 2
    DPP_MAXSTEP(x, 0x141);  // row_half_mirror     : mixes quads within 8
    DPP_MAXSTEP(x, 0x140);  // row_mirror          : mixes the two 8-groups
    return x;
}

// Load one 16-element bf16 fragment slice from a contiguous 64-float row.
// Used both for A-fragments (row = M) and for B-fragments of X^T (row = N).
__device__ __forceinline__ v16bf load_frag(const float* __restrict__ p,
                                           int kc, int hf, float scale) {
    const float4* a = (const float4*)(p + kc * 32 + 8 * hf);
    const float4* c = (const float4*)(p + kc * 32 + 16 + 8 * hf);
    float4 x0 = a[0], x1 = a[1], x2 = c[0], x3 = c[1];
    v16bf f;
    f[0] = (__bf16)(x0.x * scale); f[1] = (__bf16)(x0.y * scale);
    f[2] = (__bf16)(x0.z * scale); f[3] = (__bf16)(x0.w * scale);
    f[4] = (__bf16)(x1.x * scale); f[5] = (__bf16)(x1.y * scale);
    f[6] = (__bf16)(x1.z * scale); f[7] = (__bf16)(x1.w * scale);
    f[8]  = (__bf16)(x2.x * scale); f[9]  = (__bf16)(x2.y * scale);
    f[10] = (__bf16)(x2.z * scale); f[11] = (__bf16)(x2.w * scale);
    f[12] = (__bf16)(x3.x * scale); f[13] = (__bf16)(x3.y * scale);
    f[14] = (__bf16)(x3.z * scale); f[15] = (__bf16)(x3.w * scale);
    return f;
}

__global__ void __launch_bounds__(512, 1)
flash_attn_gfx1250(const float* __restrict__ q,
                   const float* __restrict__ k,
                   const float* __restrict__ v,
                   const float* __restrict__ mask,
                   float* __restrict__ out) {
    // mask tile: [q-row 16][k-col 32][head 16 pad 17] -> conflict-free reads
    __shared__ float  lds_mask[16][32][17];           // 34,816 B
    // probs: per-wave [row 16][col 32 pad 40] bf16 (rows 80B => 16B aligned)
    __shared__ __bf16 lds_p[16][16][40];              // 20,480 B

    const int qt   = blockIdx.x;        // 0..63 : q tile of 16 rows
    const int b    = blockIdx.y;        // 0..3  : batch
    const int tid  = threadIdx.x;       // 0..511
    const int w    = tid >> 5;          // wave index == head 0..15
    const int lane = tid & 31;
    const int l15  = lane & 15;
    const int hf   = lane >> 4;         // lane half (fragment layout)

    const int    qbase    = qt * 16;
    const size_t head_off = (size_t)b * 1024 + (size_t)w * 64;

    // ---- Q as A-fragments (pre-scaled by 1/sqrt(D)), K split 0..31 / 32..63
    const float* qp = q + (size_t)(qbase + l15) * ROW_STR + head_off;
    v16bf qa0 = load_frag(qp, 0, hf, NORM_F);
    v16bf qa1 = load_frag(qp, 1, hf, NORM_F);

    // ---- all-ones B fragment: row-sum of P via WMMA (every lane gets the sum)
    v16bf vones;
    #pragma unroll
    for (int j = 0; j < 16; ++j) vones[j] = (__bf16)1.0f;

    // ---- flash state: per-lane rows are (r + 8*hf), r = 0..7
    float rowmax[8];
    v8f   oacc[4], lacc;                // lacc[r] == running row sum
    #pragma unroll
    for (int r = 0; r < 8; ++r) rowmax[r] = -1.0e30f;
    #pragma unroll
    for (int d = 0; d < 4; ++d) oacc[d] = {};
    lacc = {};

    const int li = tid >> 5;            // staging: q-row of this thread's line
    const int lj = tid & 31;            //          k-col of this thread's line
    const float* mline = mask + (size_t)(qbase + li) * MASK_QS
                              + (size_t)lj * 64 + (size_t)b * 16;

    // ---- VGPR double buffer: g* always holds the NEXT tile's mask line
    float4 g0 = ((const float4*)mline)[0];
    float4 g1 = ((const float4*)mline)[1];
    float4 g2 = ((const float4*)mline)[2];
    float4 g3 = ((const float4*)mline)[3];

    for (int ks = 0; ks < 32; ++ks) {
        const int kb0 = ks * 32;

        __syncthreads();  // previous tile's LDS mask fully consumed

        // ---- commit prefetched mask tile to LDS (64 B contiguous per line)
        {
            float* d = &lds_mask[li][lj][0];
            d[0]=g0.x;  d[1]=g0.y;  d[2]=g0.z;  d[3]=g0.w;
            d[4]=g1.x;  d[5]=g1.y;  d[6]=g1.z;  d[7]=g1.w;
            d[8]=g2.x;  d[9]=g2.y;  d[10]=g2.z; d[11]=g2.w;
            d[12]=g3.x; d[13]=g3.y; d[14]=g3.z; d[15]=g3.w;
        }

        __syncthreads();  // mask tile visible to all waves

        // ---- issue NEXT tile's global mask loads: in flight during compute
        if (ks + 1 < 32) {
            const float* mp = mline + (size_t)(ks + 1) * 2048;
            g0 = ((const float4*)mp)[0];
            g1 = ((const float4*)mp)[1];
            g2 = ((const float4*)mp)[2];
            g3 = ((const float4*)mp)[3];
            if (ks + 2 < 32) __builtin_prefetch(mp + 2048, 0, 0);
        }

        // ---- QK^T : two 16x16 score tiles (k cols kb0+0..15, kb0+16..31)
        const float* kp0 = k + (size_t)(kb0 + l15) * ROW_STR + head_off;
        const float* kp1 = kp0 + (size_t)16 * ROW_STR;
        v16bf kb00 = load_frag(kp0, 0, hf, 1.0f);
        v16bf kb01 = load_frag(kp0, 1, hf, 1.0f);
        v16bf kb10 = load_frag(kp1, 0, hf, 1.0f);
        v16bf kb11 = load_frag(kp1, 1, hf, 1.0f);

        v8f s0 = {}, s1 = {};
        s0 = __builtin_amdgcn_wmma_f32_16x16x32_bf16(false, qa0, false, kb00,
                                                     (short)0, s0, false, false);
        s0 = __builtin_amdgcn_wmma_f32_16x16x32_bf16(false, qa1, false, kb01,
                                                     (short)0, s0, false, false);
        s1 = __builtin_amdgcn_wmma_f32_16x16x32_bf16(false, qa0, false, kb10,
                                                     (short)0, s1, false, false);
        s1 = __builtin_amdgcn_wmma_f32_16x16x32_bf16(false, qa1, false, kb11,
                                                     (short)0, s1, false, false);

        // ---- mask add + online softmax (row = r + 8*hf, col = lane group)
        #pragma unroll
        for (int r = 0; r < 8; ++r) {
            const int qi = r + 8 * hf;
            s0[r] += (lds_mask[qi][l15][w]      - 1.0f) * 10000.0f;
            s1[r] += (lds_mask[qi][16 + l15][w] - 1.0f) * 10000.0f;

            float mx    = rowmax16(fmaxf(s0[r], s1[r]));
            float mnew  = fmaxf(rowmax[r], mx);
            float scale = __expf(rowmax[r] - mnew);
            rowmax[r] = mnew;

            float p0 = __expf(s0[r] - mnew);
            float p1 = __expf(s1[r] - mnew);

            lacc[r]    *= scale;
            oacc[0][r] *= scale; oacc[1][r] *= scale;
            oacc[2][r] *= scale; oacc[3][r] *= scale;

            // write probs (C layout) into wave-private LDS (A layout source)
            lds_p[w][qi][l15]      = (__bf16)p0;
            lds_p[w][qi][16 + l15] = (__bf16)p1;
        }

        // same-wave DS ops are in-order; just stop the compiler reordering.
        asm volatile("" ::: "memory");

        // ---- read probs back as a 16x32 A-fragment
        v8bf lo = *(const v8bf*)&lds_p[w][l15][8 * hf];
        v8bf hi = *(const v8bf*)&lds_p[w][l15][16 + 8 * hf];
        v16bf pa;
        #pragma unroll
        for (int j = 0; j < 8; ++j) { pa[j] = lo[j]; pa[j + 8] = hi[j]; }

        // ---- row sums: P x ones; every lane's lacc[r] = sum_k P[row][k]
        lacc = __builtin_amdgcn_wmma_f32_16x16x32_bf16(
            false, pa, false, vones, (short)0, lacc, false, false);

        // ---- P.V : 4 d-tiles of 16 cols; V as B-fragment (coalesced rows)
        #pragma unroll
        for (int dt = 0; dt < 4; ++dt) {
            const float* vp = v + head_off + (size_t)dt * 16 + l15;
            v16bf vb;
            #pragma unroll
            for (int j = 0; j < 16; ++j)
                vb[j] = (__bf16)vp[(size_t)(kb0 + koff(hf, j)) * ROW_STR];
            oacc[dt] = __builtin_amdgcn_wmma_f32_16x16x32_bf16(
                false, pa, false, vb, (short)0, oacc[dt], false, false);
        }
    }

    // ---- epilogue: normalize and write out[t][b][h*64 + d]
    #pragma unroll
    for (int r = 0; r < 8; ++r) {
        const float inv = 1.0f / lacc[r];
        const int   t   = qbase + r + 8 * hf;
        float* op = out + (size_t)t * ROW_STR + head_off;
        op[0 * 16 + l15] = oacc[0][r] * inv;
        op[1 * 16 + l15] = oacc[1][r] * inv;
        op[2 * 16 + l15] = oacc[2][r] * inv;
        op[3 * 16 + l15] = oacc[3][r] * inv;
    }
}

extern "C" void kernel_launch(void* const* d_in, const int* in_sizes, int n_in,
                              void* d_out, int out_size, void* d_ws, size_t ws_size,
                              hipStream_t stream) {
    const float* q  = (const float*)d_in[0];
    const float* k  = (const float*)d_in[1];
    const float* v  = (const float*)d_in[2];
    const float* m  = (const float*)d_in[3];
    float* out = (float*)d_out;
    (void)in_sizes; (void)n_in; (void)out_size; (void)d_ws; (void)ws_size;

    dim3 grid(64, 4, 1);     // 64 q-tiles x 4 batches; 16 waves (=16 heads) each
    dim3 block(512, 1, 1);
    hipLaunchKernelGGL(flash_attn_gfx1250, grid, block, 0, stream, q, k, v, m, out);
}
